// DHSRNN_6038724018535
// MI455X (gfx1250) — matrix operand order repaired and checked by reading, not compiled, not run on hardware
//
#include <hip/hip_runtime.h>
#include <math.h>

// ---------------- types ----------------
typedef __bf16 bf16_t;
typedef __bf16 v8bf  __attribute__((ext_vector_type(8)));
typedef __bf16 v16bf __attribute__((ext_vector_type(16)));
typedef float  v8f   __attribute__((ext_vector_type(8)));

// ---------------- problem constants ----------------
#define IN_DIMC 700
#define HIDDENC 1024
#define OUT_DIMC 20
#define BRANCHC 4
#define BATCHC 128
#define TLENC 250
#define ISZC   (IN_DIMC + HIDDENC)   // 1724
#define KPC    1728                  // K padded to 54*32
#define NKC    (KPC / 32)            // 54
#define HBC    (HIDDENC * BRANCHC)   // 4096
#define VTHC   1.0f
#define NBLK   128
#define NTHR   256

// ---------------- workspace layout (bytes, 256B aligned) ----------------
static constexpr size_t OFF_WBF    = 0;                                      // bf16 [4096][1728]
static constexpr size_t OFF_WRBF   = OFF_WBF    + (size_t)HBC * KPC * 2;     // bf16 [32][1024]
static constexpr size_t OFF_KBUF   = OFF_WRBF   + (size_t)32 * HIDDENC * 2;  // bf16 [128][1728]
static constexpr size_t OFF_SPKBF  = OFF_KBUF   + (size_t)BATCHC * KPC * 2;  // bf16 [128][1024]
static constexpr size_t OFF_DIN    = OFF_SPKBF  + (size_t)BATCHC * HIDDENC * 2; // f32 [128][4096]
static constexpr size_t OFF_MEM    = OFF_DIN    + (size_t)BATCHC * HBC * 4;  // f32 [128][1024]
static constexpr size_t OFF_RMEM   = OFF_MEM    + (size_t)BATCHC * HIDDENC * 4; // f32 [128][32]
static constexpr size_t OFF_ACC    = OFF_RMEM   + (size_t)BATCHC * 32 * 4;   // f32 [128][20]
static constexpr size_t OFF_BETA   = OFF_ACC    + (size_t)BATCHC * OUT_DIMC * 4; // f32 [4096]
static constexpr size_t OFF_ALPHA  = OFF_BETA   + (size_t)HBC * 4;           // f32 [1024]
static constexpr size_t OFF_ALPHAR = OFF_ALPHA  + (size_t)HIDDENC * 4;       // f32 [32] (pad 256)
static constexpr size_t OFF_CNT    = OFF_ALPHAR + 256;                       // u32 barrier counter

// ---------------- helpers ----------------
__device__ __forceinline__ float sigm(float x) { return 1.0f / (1.0f + __expf(-x)); }

// A-fragment (16x32 bf16): lane 0-15 -> row M, K 0..15; lane 16-31 -> K 16..31
// caller passes p = rowptr + kk*32 + half*16 ; elements [0..15]=p[0..15]
__device__ __forceinline__ v16bf lda(const bf16_t* p) {
  v8bf lo = *reinterpret_cast<const v8bf*>(p);
  v8bf hi = *reinterpret_cast<const v8bf*>(p + 8);
  v16bf f;
#pragma unroll
  for (int i = 0; i < 8; ++i) { f[i] = lo[i]; f[i + 8] = hi[i]; }
  return f;
}

// B-fragment (32x16 bf16): lane 0-15 -> col N=lane, K 0..15 ; lane 16-31 -> K 16..31
// caller passes p = colrowptr + kk*32 + half*16 ; 16 contiguous elements
__device__ __forceinline__ v16bf ldb(const bf16_t* p) {
  v8bf lo = *reinterpret_cast<const v8bf*>(p);
  v8bf hi = *reinterpret_cast<const v8bf*>(p + 8);
  v16bf f;
#pragma unroll
  for (int i = 0; i < 8; ++i) { f[i] = lo[i]; f[i + 8] = hi[i]; }
  return f;
}

__device__ __forceinline__ v8f wmma_bf16(v16bf a, v16bf b, v8f c) {
  // D = A*B + C, f32 accumulate -> v_wmma_f32_16x16x32_bf16
  return __builtin_amdgcn_wmma_f32_16x16x32_bf16(false, a, false, b, (short)0, c, false, false);
}

// grid-wide barrier: monotonic counter, reset by prep kernel each launch
__device__ __forceinline__ void gbar(unsigned* cnt, unsigned target) {
  __threadfence();
  __syncthreads();
  if (threadIdx.x == 0) {
    __hip_atomic_fetch_add(cnt, 1u, __ATOMIC_ACQ_REL, __HIP_MEMORY_SCOPE_AGENT);
    while (__hip_atomic_load(cnt, __ATOMIC_ACQUIRE, __HIP_MEMORY_SCOPE_AGENT) < target)
      __builtin_amdgcn_s_sleep(2);
  }
  __syncthreads();
}

// ---------------- prep kernel: pack weights to bf16, init state ----------------
__global__ void __launch_bounds__(NTHR) dhsnn_prep(
    const float* __restrict__ W, const float* __restrict__ Wr,
    const float* __restrict__ tau_m, const float* __restrict__ tau_n,
    const float* __restrict__ tau_r, const float* __restrict__ mem0,
    bf16_t* __restrict__ wbf, bf16_t* __restrict__ wrbf,
    bf16_t* __restrict__ kbuf, bf16_t* __restrict__ spkbf,
    float* __restrict__ din, float* __restrict__ memst,
    float* __restrict__ rmem, float* __restrict__ acc,
    float* __restrict__ beta, float* __restrict__ alpha,
    float* __restrict__ alphar, unsigned* __restrict__ cnt)
{
  const size_t gid = (size_t)blockIdx.x * blockDim.x + threadIdx.x;
  const size_t gs  = (size_t)gridDim.x * blockDim.x;
  if (gid == 0) *cnt = 0u;
  for (size_t i = gid; i < (size_t)HBC * KPC; i += gs) {
    size_t n = i / KPC, k = i - n * KPC;
    wbf[i] = (bf16_t)((k < ISZC) ? W[n * ISZC + k] : 0.0f);
  }
  for (size_t i = gid; i < (size_t)32 * HIDDENC; i += gs) {
    size_t n = i / HIDDENC, k = i - n * HIDDENC;
    wrbf[i] = (bf16_t)((n < OUT_DIMC) ? Wr[n * HIDDENC + k] : 0.0f);
  }
  for (size_t i = gid; i < (size_t)BATCHC * KPC; i += gs)      kbuf[i]  = (bf16_t)0.0f;
  for (size_t i = gid; i < (size_t)BATCHC * HIDDENC; i += gs)  spkbf[i] = (bf16_t)0.0f;
  for (size_t i = gid; i < (size_t)BATCHC * HBC; i += gs)      din[i]   = 0.0f;
  for (size_t i = gid; i < (size_t)BATCHC * HIDDENC; i += gs)  memst[i] = mem0[i];
  for (size_t i = gid; i < (size_t)BATCHC * 32; i += gs)       rmem[i]  = 0.0f;
  for (size_t i = gid; i < (size_t)BATCHC * OUT_DIMC; i += gs) acc[i]   = 0.0f;
  for (size_t i = gid; i < (size_t)HBC; i += gs)               beta[i]  = sigm(tau_n[i]);
  for (size_t i = gid; i < (size_t)HIDDENC; i += gs)           alpha[i] = sigm(tau_m[i]);
  for (size_t i = gid; i < 32; i += gs) alphar[i] = (i < OUT_DIMC) ? sigm(tau_r[i]) : 0.0f;
}

// ---------------- persistent timestep kernel ----------------
// NOTE: wbf/wrbf are deliberately NOT __restrict__ so weight loads cannot be
// hoisted (LICM) across the t loop and spilled to scratch; they must reissue
// from L2 each timestep.
__global__ void __launch_bounds__(NTHR) dhsnn_main(
    const float* __restrict__ x, const float* __restrict__ bvec,
    const float* __restrict__ brv,
    const bf16_t* wbf, const bf16_t* wrbf,
    bf16_t* __restrict__ kbuf, bf16_t* __restrict__ spkbf,
    float* __restrict__ din, float* __restrict__ memst,
    float* __restrict__ rmem, float* __restrict__ acc,
    const float* __restrict__ beta, const float* __restrict__ alpha,
    const float* __restrict__ alphar,
    unsigned* __restrict__ cnt, float* __restrict__ out)
{
  const int tid  = threadIdx.x;
  const int lane = tid & 31;
  const int lr   = lane & 15;
  const int half = lane >> 4;

  // tile assignment: 1024 waves, each owns tiles (bm0,bn) and (bm0+1,bn), sharing B
  const int wgl = blockIdx.x * (NTHR / 32) + (tid >> 5);   // 0..1023
  const int bn  = wgl & 255;                               // N tile 0..255
  const int bm0 = (wgl >> 8) * 2;                          // M tile pair base {0,2,4,6}

  const int   n      = bn * 16 + lr;                       // this lane's output column
  const int   h      = n >> 2;                             // neuron
  const float bias_n = bvec[n];
  const float beta_n = beta[n];
  const float al_h   = alpha[h];
  const float one_mb = 1.0f - beta_n;
  const float one_ma = 1.0f - al_h;

  const bf16_t* arow0 = kbuf + (size_t)(bm0 * 16 + lr) * KPC;
  const bf16_t* arow1 = arow0 + (size_t)16 * KPC;
  const bf16_t* browp = wbf + (size_t)n * KPC;

  unsigned target = 0;

  for (int t = 0; t < TLENC; ++t) {
    // ---- phase 1: refresh x columns of the K-buffer (bf16) ----
    // also prefetch next timestep's x slice (the only steady-state HBM traffic)
    for (int idx = blockIdx.x * NTHR + tid; idx < BATCHC * IN_DIMC; idx += NBLK * NTHR) {
      int b = idx / IN_DIMC, i = idx - b * IN_DIMC;
      const float* xp = x + ((size_t)b * TLENC + t) * IN_DIMC + i;
      kbuf[(size_t)b * KPC + i] = (bf16_t)(*xp);
      if (t + 1 < TLENC) __builtin_prefetch(xp + IN_DIMC, 0, 1);  // global_prefetch_b8
    }
    target += gridDim.x;
    gbar(cnt, target);

    // ---- phase 2: WMMA GEMM + dendrite + soma + spike (fully fused per tile) ----
    v8f c0 = {}, c1 = {};
#pragma unroll 2
    for (int kk = 0; kk < NKC; ++kk) {
      const int ka = kk * 32 + half * 16;
      const int kb = kk * 32 + half * 16;
      v16bf bf = ldb(browp + kb);
      v16bf a0 = lda(arow0 + ka);
      v16bf a1 = lda(arow1 + ka);
      c0 = wmma_bf16(a0, bf, c0);
      c1 = wmma_bf16(a1, bf, c1);
    }

#pragma unroll
    for (int r = 0; r < 8; ++r) {
      // tile bm0 : batch rows bm0*16 + half*8 + r
      {
        const int b = bm0 * 16 + half * 8 + r;
        const size_t di = (size_t)b * HBC + n;
        float cur = c0[r] + bias_n;
        float dn  = beta_n * din[di] + one_mb * cur;
        din[di] = dn;
        float s4 = dn + __shfl_xor(dn, 1, 32);
        s4 += __shfl_xor(s4, 2, 32);
        if ((lr & 3) == 0) {
          const size_t mi = (size_t)b * HIDDENC + h;
          float m  = memst[mi];
          float sp = (float)spkbf[mi];
          m = m * al_h + one_ma * s4 - VTHC * sp;
          float sn = (m > VTHC) ? 1.0f : 0.0f;
          memst[mi] = m;
          bf16_t sb = (bf16_t)sn;
          spkbf[mi] = sb;
          kbuf[(size_t)b * KPC + IN_DIMC + h] = sb;
        }
      }
      // tile bm0+1
      {
        const int b = (bm0 + 1) * 16 + half * 8 + r;
        const size_t di = (size_t)b * HBC + n;
        float cur = c1[r] + bias_n;
        float dn  = beta_n * din[di] + one_mb * cur;
        din[di] = dn;
        float s4 = dn + __shfl_xor(dn, 1, 32);
        s4 += __shfl_xor(s4, 2, 32);
        if ((lr & 3) == 0) {
          const size_t mi = (size_t)b * HIDDENC + h;
          float m  = memst[mi];
          float sp = (float)spkbf[mi];
          m = m * al_h + one_ma * s4 - VTHC * sp;
          float sn = (m > VTHC) ? 1.0f : 0.0f;
          memst[mi] = m;
          bf16_t sb = (bf16_t)sn;
          spkbf[mi] = sb;
          kbuf[(size_t)b * KPC + IN_DIMC + h] = sb;
        }
      }
    }
    target += gridDim.x;
    gbar(cnt, target);

    // ---- phase 3 (block 0 only): WMMA readout + leaky integrate + softmax acc ----
    if (blockIdx.x == 0) {
      const int wv = tid >> 5;                                  // M tile = 0..7
      const bf16_t* ar  = spkbf + (size_t)(wv * 16 + lr) * HIDDENC;
      const bf16_t* br0 = wrbf + (size_t)lr * HIDDENC;          // cols 0..15
      const bf16_t* br1 = wrbf + (size_t)(16 + lr) * HIDDENC;   // cols 16..31
      v8f r0 = {}, r1 = {};
#pragma unroll 2
      for (int kk = 0; kk < HIDDENC / 32; ++kk) {
        const int ka = kk * 32 + half * 16;
        const int kb = kk * 32 + half * 16;
        v16bf a = lda(ar + ka);
        r0 = wmma_bf16(a, ldb(br0 + kb), r0);
        r1 = wmma_bf16(a, ldb(br1 + kb), r1);
      }
#pragma unroll
      for (int r = 0; r < 8; ++r) {
        const int b = wv * 16 + half * 8 + r;
        {   // n0 = lr (always < 20)
          const float arf = alphar[lr];
          const size_t ri = (size_t)b * 32 + lr;
          rmem[ri] = arf * rmem[ri] + (1.0f - arf) * (r0[r] + brv[lr]);
        }
        const int n1 = 16 + lr;
        if (n1 < OUT_DIMC) {
          const float arf = alphar[n1];
          const size_t ri = (size_t)b * 32 + n1;
          rmem[ri] = arf * rmem[ri] + (1.0f - arf) * (r1[r] + brv[n1]);
        }
      }
      __syncthreads();
      if (tid < BATCHC && t > 0) {           // WARMUP==0: accumulate for t >= 1
        float vals[OUT_DIMC];
        float mx = -1e30f;
#pragma unroll
        for (int o = 0; o < OUT_DIMC; ++o) {
          vals[o] = rmem[(size_t)tid * 32 + o];
          mx = fmaxf(mx, vals[o]);
        }
        float s = 0.0f;
#pragma unroll
        for (int o = 0; o < OUT_DIMC; ++o) { vals[o] = __expf(vals[o] - mx); s += vals[o]; }
        const float inv = 1.0f / s;
#pragma unroll
        for (int o = 0; o < OUT_DIMC; ++o) acc[(size_t)tid * OUT_DIMC + o] += vals[o] * inv;
      }
      __syncthreads();
    }
  }

  // ---- epilogue: block 0 owns acc, copy to output ----
  if (blockIdx.x == 0) {
    __syncthreads();
    for (int i = tid; i < BATCHC * OUT_DIMC; i += NTHR) out[i] = acc[i];
  }
}

// ---------------- host entry ----------------
extern "C" void kernel_launch(void* const* d_in, const int* in_sizes, int n_in,
                              void* d_out, int out_size, void* d_ws, size_t ws_size,
                              hipStream_t stream) {
  (void)in_sizes; (void)n_in; (void)out_size; (void)ws_size;
  const float* x     = (const float*)d_in[0];
  const float* W     = (const float*)d_in[1];
  const float* bvec  = (const float*)d_in[2];
  const float* tau_m = (const float*)d_in[3];
  const float* tau_n = (const float*)d_in[4];
  const float* Wr    = (const float*)d_in[5];
  const float* brv   = (const float*)d_in[6];
  const float* tau_r = (const float*)d_in[7];
  const float* mem0  = (const float*)d_in[8];
  float* out = (float*)d_out;

  char* ws = (char*)d_ws;
  bf16_t*   wbf    = (bf16_t*)(ws + OFF_WBF);
  bf16_t*   wrbf   = (bf16_t*)(ws + OFF_WRBF);
  bf16_t*   kbuf   = (bf16_t*)(ws + OFF_KBUF);
  bf16_t*   spkbf  = (bf16_t*)(ws + OFF_SPKBF);
  float*    din    = (float*)(ws + OFF_DIN);
  float*    memst  = (float*)(ws + OFF_MEM);
  float*    rmem   = (float*)(ws + OFF_RMEM);
  float*    acc    = (float*)(ws + OFF_ACC);
  float*    beta   = (float*)(ws + OFF_BETA);
  float*    alpha  = (float*)(ws + OFF_ALPHA);
  float*    alphar = (float*)(ws + OFF_ALPHAR);
  unsigned* cnt    = (unsigned*)(ws + OFF_CNT);

  dhsnn_prep<<<NBLK, NTHR, 0, stream>>>(W, Wr, tau_m, tau_n, tau_r, mem0,
                                        wbf, wrbf, kbuf, spkbf, din, memst,
                                        rmem, acc, beta, alpha, alphar, cnt);
  dhsnn_main<<<NBLK, NTHR, 0, stream>>>(x, bvec, brv, wbf, wrbf, kbuf, spkbf,
                                        din, memst, rmem, acc, beta, alpha,
                                        alphar, cnt, out);
}
